// HunyuanVideoSingleStreamBlock_72559177499129
// MI455X (gfx1250) — matrix operand order repaired and checked by reading, not compiled, not run on hardware
//
#include <hip/hip_runtime.h>
#include <hip/hip_bf16.h>
#include <math.h>

// ---------------- problem constants ----------------
#define S_LEN    2304
#define D_MODEL  3072
#define NHEAD    24
#define HDIM     128
#define IMG_LEN  2048
#define QKV_DIM  9216    // 3*D
#define PROJ_DIM 21504   // 3*D + 4*D
#define MLP_DIM  12288
#define K2_DIM   15360   // D + MLP
#define MOD_DIM  9216

typedef __bf16 bf16;
typedef __attribute__((ext_vector_type(8)))  __bf16 v8bf;
typedef __attribute__((ext_vector_type(16))) __bf16 v16bf;
typedef __attribute__((ext_vector_type(8)))  float  v8f;
typedef __attribute__((ext_vector_type(4)))  float  v4f;
typedef int v4i __attribute__((vector_size(16)));   // matches builtin param type

// ---------------- async global->LDS copy (CDNA5 ASYNCcnt path) ----------------
#if defined(__has_builtin)
#  if __has_builtin(__builtin_amdgcn_global_load_async_to_lds_b128)
#    define USE_ASYNC_LDS 1
#  endif
#endif
#ifndef USE_ASYNC_LDS
#  define USE_ASYNC_LDS 0
#endif

__device__ inline void cp_b128(void* lds_dst, const void* gsrc) {
#if USE_ASYNC_LDS
  __builtin_amdgcn_global_load_async_to_lds_b128(
      (__attribute__((address_space(1))) v4i*)gsrc,
      (__attribute__((address_space(3))) v4i*)lds_dst, 0, 0);
#else
  *(v8bf*)lds_dst = *(const v8bf*)gsrc;
#endif
}

__device__ inline void wait_async_lds() {
#if USE_ASYNC_LDS
#  if defined(__has_builtin) && __has_builtin(__builtin_amdgcn_s_wait_asynccnt)
  __builtin_amdgcn_s_wait_asynccnt(0);
#  else
  asm volatile("s_wait_asynccnt 0" ::: "memory");
#  endif
#endif
}

// ---------------- WMMA helpers (CDNA5 wave32 layouts) ----------------
// A fragment: 16(M)x32(K) bf16. lanes 0-15 hold M=lane, K {0..7,16..23};
// lanes 16-31 hold M=lane-16, K {8..15,24..31}. LDS tile is row-major [m][k].
__device__ inline v16bf load_frag_a(const bf16* lds, int rowbase, int stride,
                                    int kbase, int lane) {
  int khalf = lane >> 4;
  int m     = lane & 15;
  const bf16* p = lds + (size_t)(rowbase + m) * stride + kbase + khalf * 8;
  union { v16bf v; v8bf h[2]; } u;
  u.h[0] = *(const v8bf*)(p);
  u.h[1] = *(const v8bf*)(p + 16);
  return u.v;
}

// B fragment: 32(K)x16(N) bf16. lanes 0-15: N=lane, K {0..7,16..23};
// lanes 16-31: N=lane-16, K {8..15,24..31}. LDS tile is [n][k] row-major.
__device__ inline v16bf load_frag_b(const bf16* lds, int colbase, int stride,
                                    int kbase, int lane) {
  int khalf = lane >> 4;
  int n     = lane & 15;
  const bf16* p = lds + (size_t)(colbase + n) * stride + kbase + khalf * 8;
  union { v16bf v; v8bf h[2]; } u;
  u.h[0] = *(const v8bf*)(p);
  u.h[1] = *(const v8bf*)(p + 16);
  return u.v;
}

__device__ inline v8f wmma_bf16(v16bf a, v16bf b, v8f c) {
  return __builtin_amdgcn_wmma_f32_16x16x32_bf16(false, a, false, b,
                                                 (short)0, c, false, false);
}

__device__ inline float gelu_tanh(float x) {
  float x3 = x * x * x;
  return 0.5f * x * (1.0f + tanhf(0.7978845608028654f * (x + 0.044715f * x3)));
}

// ------------- kernel 0: convert+transpose fp32 W[K][N] -> bf16 WT[N][K] -----
__global__ __launch_bounds__(256)
void convt_kernel(const float* __restrict__ W, bf16* __restrict__ WT,
                  int K, int N) {
  __shared__ float tile[32][33];
  int bx = blockIdx.x;            // tile along N
  int by = blockIdx.y;            // tile along K
  int tid = threadIdx.x;
  int r = tid >> 5, c = tid & 31; // 8 rows x 32 cols per pass
#pragma unroll
  for (int rr = 0; rr < 4; ++rr) {
    int k = by * 32 + r + rr * 8;
    tile[r + rr * 8][c] = W[(size_t)k * N + bx * 32 + c];
  }
  __syncthreads();
#pragma unroll
  for (int rr = 0; rr < 4; ++rr) {
    int n = bx * 32 + r + rr * 8;
    WT[(size_t)n * K + by * 32 + c] = (bf16)tile[c][r + rr * 8];
  }
}

// ---------------- kernel 1: mod = silu(vec) @ mod_w + mod_b ----------------
__global__ __launch_bounds__(256)
void mod_gemv_kernel(const float* __restrict__ vec,
                     const float* __restrict__ mod_w,
                     const float* __restrict__ mod_b,
                     float* __restrict__ mod) {
  __shared__ float sv[D_MODEL];
  int tid = threadIdx.x;
  for (int i = tid; i < D_MODEL; i += 256) {
    float v = vec[i];
    sv[i] = v / (1.0f + __expf(-v));
  }
  __syncthreads();
  int col = blockIdx.x * 256 + tid;
  float acc = mod_b[col];
  for (int i = 0; i < D_MODEL; ++i)
    acc = fmaf(sv[i], mod_w[(size_t)i * MOD_DIM + col], acc);
  mod[col] = acc;
}

// ---------------- kernel 2: LayerNorm + modulate -> bf16 xm ----------------
__global__ __launch_bounds__(256)
void ln_mod_kernel(const float* __restrict__ x,
                   const float* __restrict__ mod,
                   bf16* __restrict__ xm) {
  int row = blockIdx.x, tid = threadIdx.x;
  const float* xr = x + (size_t)row * D_MODEL;
  float lv[12];
  float s = 0.f, ss = 0.f;
#pragma unroll
  for (int i = 0; i < 12; ++i) {
    float v = xr[tid + i * 256];
    lv[i] = v; s += v; ss += v * v;
  }
  for (int off = 16; off > 0; off >>= 1) {
    s  += __shfl_down(s,  off, 32);
    ss += __shfl_down(ss, off, 32);
  }
  __shared__ float ps[8], pss[8];
  int wid = tid >> 5, lane = tid & 31;
  if (lane == 0) { ps[wid] = s; pss[wid] = ss; }
  __syncthreads();
  if (tid == 0) {
    float a = 0.f, b = 0.f;
    for (int w = 0; w < 8; ++w) { a += ps[w]; b += pss[w]; }
    ps[0] = a; pss[0] = b;
  }
  __syncthreads();
  float mu  = ps[0]  * (1.0f / D_MODEL);
  float var = pss[0] * (1.0f / D_MODEL) - mu * mu;
  float rs  = rsqrtf(var + 1e-6f);
  const float* shift = mod;
  const float* scale = mod + D_MODEL;
  bf16* out = xm + (size_t)row * D_MODEL;
#pragma unroll
  for (int i = 0; i < 12; ++i) {
    int c = tid + i * 256;
    float v = (lv[i] - mu) * rs;
    out[c] = (bf16)(v * (1.0f + scale[c]) + shift[c]);
  }
}

// ---------------- kernel 3: GEMM1  proj = xm @ w1 + b1 ----------------
// 128x128 block tile, BK=32, 8 waves (4Mx2N). Double-buffered async LDS copies.
__global__ __launch_bounds__(256)
void gemm1_kernel(const bf16* __restrict__ A, const bf16* __restrict__ WT,
                  const float* __restrict__ bias,
                  float* __restrict__ projq, bf16* __restrict__ a2) {
  __shared__ __align__(16) bf16 As[2][128 * 32];  // [m][k]
  __shared__ __align__(16) bf16 Bs[2][128 * 32];  // [n][k]
  int tid = threadIdx.x, lane = tid & 31, wid = tid >> 5;
  int waveM = wid & 3, waveN = wid >> 2;
  int m0 = blockIdx.y * 128, n0 = blockIdx.x * 128;

  auto issue = [&](int buf, int k0) {
#pragma unroll
    for (int i = tid * 8; i < 128 * 32; i += 2048) {
      int r = i >> 5, c = i & 31;
      cp_b128(&As[buf][i], &A [(size_t)(m0 + r) * D_MODEL + k0 + c]);
      cp_b128(&Bs[buf][i], &WT[(size_t)(n0 + r) * D_MODEL + k0 + c]);
    }
  };

  v8f z = {0.f,0.f,0.f,0.f,0.f,0.f,0.f,0.f};
  v8f acc[2][4];
#pragma unroll
  for (int mi = 0; mi < 2; ++mi)
#pragma unroll
    for (int ni = 0; ni < 4; ++ni) acc[mi][ni] = z;

  issue(0, 0);
  for (int k0 = 0; k0 < D_MODEL; k0 += 32) {
    int buf = (k0 >> 5) & 1;
    wait_async_lds();
    __syncthreads();
    if (k0 + 32 < D_MODEL) issue(buf ^ 1, k0 + 32);  // overlap with compute

    v16bf afr[2], bfr[4];
#pragma unroll
    for (int mi = 0; mi < 2; ++mi)
      afr[mi] = load_frag_a(&As[buf][0], waveM * 32 + mi * 16, 32, 0, lane);
#pragma unroll
    for (int ni = 0; ni < 4; ++ni)
      bfr[ni] = load_frag_b(&Bs[buf][0], waveN * 64 + ni * 16, 32, 0, lane);
#pragma unroll
    for (int mi = 0; mi < 2; ++mi)
#pragma unroll
      for (int ni = 0; ni < 4; ++ni)
        acc[mi][ni] = wmma_bf16(afr[mi], bfr[ni], acc[mi][ni]);
  }

  int halfoff = (lane >> 4) * 8;
#pragma unroll
  for (int mi = 0; mi < 2; ++mi)
#pragma unroll
    for (int ni = 0; ni < 4; ++ni) {
      int col = n0 + waveN * 64 + ni * 16 + (lane & 15);
      float bv = bias[col];
#pragma unroll
      for (int e = 0; e < 8; ++e) {
        int row = m0 + waveM * 32 + mi * 16 + e + halfoff;
        float v = acc[mi][ni][e] + bv;
        if (col < QKV_DIM) {
          projq[(size_t)row * QKV_DIM + col] = v;
        } else {
          a2[(size_t)row * K2_DIM + D_MODEL + (col - QKV_DIM)] = (bf16)gelu_tanh(v);
        }
      }
    }
}

// -------- kernel 4: qkv split + RMSNorm + RoPE -> q/k [h][s][hd], v [h][hd][s]
__global__ __launch_bounds__(128)
void qkv_prep_kernel(const float* __restrict__ projq,
                     const float* __restrict__ qw, const float* __restrict__ kw,
                     const float* __restrict__ fcos, const float* __restrict__ fsin,
                     bf16* __restrict__ qh, bf16* __restrict__ kh,
                     bf16* __restrict__ vt) {
  int s = blockIdx.x, h = blockIdx.y, d = threadIdx.x;
  const float* base = projq + (size_t)s * QKV_DIM + h * HDIM;
  float qv = base[d];
  float kv = base[D_MODEL + d];
  float vv = base[2 * D_MODEL + d];

  float q2 = qv * qv, k2 = kv * kv;
  for (int off = 16; off > 0; off >>= 1) {
    q2 += __shfl_down(q2, off, 32);
    k2 += __shfl_down(k2, off, 32);
  }
  __shared__ float pq[4], pk[4];
  int wid = d >> 5, lane = d & 31;
  if (lane == 0) { pq[wid] = q2; pk[wid] = k2; }
  __syncthreads();
  float qs = pq[0] + pq[1] + pq[2] + pq[3];
  float ks = pk[0] + pk[1] + pk[2] + pk[3];

  float qn = qv * rsqrtf(qs * (1.0f / HDIM) + 1e-6f) * qw[d];
  float kn = kv * rsqrtf(ks * (1.0f / HDIM) + 1e-6f) * kw[d];

  if (s < IMG_LEN) {
    float c  = fcos[(size_t)s * HDIM + d];
    float sn = fsin[(size_t)s * HDIM + d];
    float qo = __shfl_xor(qn, 1, 32);   // partner d^1 (same wave)
    float ko = __shfl_xor(kn, 1, 32);
    float sgn = (d & 1) ? 1.0f : -1.0f; // rot[2i]=-u[2i+1], rot[2i+1]=u[2i]
    qn = qn * c + sgn * qo * sn;
    kn = kn * c + sgn * ko * sn;
  }
  size_t o = ((size_t)h * S_LEN + s) * HDIM + d;
  qh[o] = (bf16)qn;
  kh[o] = (bf16)kn;
  vt[((size_t)h * HDIM + d) * S_LEN + s] = (bf16)vv;   // transposed for attention
}

// ---------------- kernel 5: flash attention -> ao into a2[:, 0:3072] --------
__global__ __launch_bounds__(256)
void attn_kernel(const bf16* __restrict__ qh, const bf16* __restrict__ kh,
                 const bf16* __restrict__ vt, bf16* __restrict__ a2) {
  __shared__ __align__(16) bf16 smA[128 * 128]; // Qs; later Ks[64][128] | VsT[128][64]
  __shared__ __align__(16) bf16 Ps[128 * 64];   // [row][key]
  bf16* Ks  = smA;
  bf16* VsT = smA + 64 * 128;

  int tid = threadIdx.x, lane = tid & 31, w = tid >> 5;
  int q0 = blockIdx.x * 128, h = blockIdx.y;
  const bf16* Qg = qh + ((size_t)h * S_LEN + q0) * HDIM;
  const bf16* Kg = kh + (size_t)h * S_LEN * HDIM;
  const bf16* Vt = vt + (size_t)h * HDIM * S_LEN;   // [d][s]

  // stage Q tile, pull this wave's A-fragments into registers
  for (int i = tid * 8; i < 128 * 128; i += 2048)
    cp_b128(&smA[i], &Qg[i]);
  wait_async_lds();
  __syncthreads();
  v16bf qf[4];
#pragma unroll
  for (int kc = 0; kc < 4; ++kc)
    qf[kc] = load_frag_a(smA, w * 16, 128, kc * 32, lane);
  __syncthreads();

  v8f z = {0.f,0.f,0.f,0.f,0.f,0.f,0.f,0.f};
  v8f o[8];
#pragma unroll
  for (int no = 0; no < 8; ++no) o[no] = z;
  float m_r[8], l_r[8];
#pragma unroll
  for (int e = 0; e < 8; ++e) { m_r[e] = -1e30f; l_r[e] = 0.f; }

  const float sc = 0.08838834764831845f; // 1/sqrt(128)

  for (int kt = 0; kt < S_LEN / 64; ++kt) {
    int kbase = kt * 64;
    __syncthreads();
    for (int i = tid * 8; i < 64 * 128; i += 2048)            // K tile [key][d]
      cp_b128(&Ks[i], &Kg[(size_t)kbase * HDIM + i]);
    for (int i = tid * 8; i < 128 * 64; i += 2048) {          // V tile [d][key]
      int r = i >> 6, c = i & 63;
      cp_b128(&VsT[i], &Vt[(size_t)r * S_LEN + kbase + c]);
    }
    wait_async_lds();
    __syncthreads();

    // S = Q . K^T  (wave strip 16 x 64)
    v8f sacc[4];
#pragma unroll
    for (int ni = 0; ni < 4; ++ni) {
      sacc[ni] = z;
#pragma unroll
      for (int kc = 0; kc < 4; ++kc)
        sacc[ni] = wmma_bf16(qf[kc],
                             load_frag_b(Ks, ni * 16, 128, kc * 32, lane),
                             sacc[ni]);
    }

    // online softmax per row (registers only; 16-lane groups)
#pragma unroll
    for (int e = 0; e < 8; ++e) {
      float mx = -1e30f;
#pragma unroll
      for (int ni = 0; ni < 4; ++ni) mx = fmaxf(mx, sacc[ni][e] * sc);
      for (int off = 1; off < 16; off <<= 1)
        mx = fmaxf(mx, __shfl_xor(mx, off, 32));
      float mnew  = fmaxf(m_r[e], mx);
      float alpha = __expf(m_r[e] - mnew);
      m_r[e] = mnew;
      float rs = 0.f;
      int prow = w * 16 + e + ((lane >> 4) << 3);
#pragma unroll
      for (int ni = 0; ni < 4; ++ni) {
        float p = __expf(sacc[ni][e] * sc - mnew);
        rs += p;
        Ps[prow * 64 + ni * 16 + (lane & 15)] = (bf16)p;
      }
      for (int off = 1; off < 16; off <<= 1)
        rs += __shfl_xor(rs, off, 32);
      l_r[e] = l_r[e] * alpha + rs;
#pragma unroll
      for (int no = 0; no < 8; ++no) o[no][e] *= alpha;
    }

    // O += P . V   (P rows are wave-private; C->A relayout via Ps)
#pragma unroll
    for (int kc = 0; kc < 2; ++kc) {
      v16bf pa = load_frag_a(Ps, w * 16, 64, kc * 32, lane);
#pragma unroll
      for (int no = 0; no < 8; ++no)
        o[no] = wmma_bf16(pa,
                          load_frag_b(VsT, no * 16, 64, kc * 32, lane),
                          o[no]);
    }
  }

  int halfoff = (lane >> 4) << 3;
#pragma unroll
  for (int no = 0; no < 8; ++no)
#pragma unroll
    for (int e = 0; e < 8; ++e) {
      int row = q0 + w * 16 + e + halfoff;
      int col = h * HDIM + no * 16 + (lane & 15);
      a2[(size_t)row * K2_DIM + col] = (bf16)(o[no][e] / l_r[e]);
    }
}

// ---------------- kernel 6: GEMM2  out = x + ([ao|mlp] @ w2 + b2) * gate ----
__global__ __launch_bounds__(256)
void gemm2_kernel(const bf16* __restrict__ A, const bf16* __restrict__ WT,
                  const float* __restrict__ bias, const float* __restrict__ mod,
                  const float* __restrict__ x, float* __restrict__ out) {
  __shared__ __align__(16) bf16 As[2][128 * 32];
  __shared__ __align__(16) bf16 Bs[2][128 * 32];
  int tid = threadIdx.x, lane = tid & 31, wid = tid >> 5;
  int waveM = wid & 3, waveN = wid >> 2;
  int m0 = blockIdx.y * 128, n0 = blockIdx.x * 128;

  auto issue = [&](int buf, int k0) {
#pragma unroll
    for (int i = tid * 8; i < 128 * 32; i += 2048) {
      int r = i >> 5, c = i & 31;
      cp_b128(&As[buf][i], &A [(size_t)(m0 + r) * K2_DIM + k0 + c]);
      cp_b128(&Bs[buf][i], &WT[(size_t)(n0 + r) * K2_DIM + k0 + c]);
    }
  };

  v8f z = {0.f,0.f,0.f,0.f,0.f,0.f,0.f,0.f};
  v8f acc[2][4];
#pragma unroll
  for (int mi = 0; mi < 2; ++mi)
#pragma unroll
    for (int ni = 0; ni < 4; ++ni) acc[mi][ni] = z;

  issue(0, 0);
  for (int k0 = 0; k0 < K2_DIM; k0 += 32) {
    int buf = (k0 >> 5) & 1;
    wait_async_lds();
    __syncthreads();
    if (k0 + 32 < K2_DIM) issue(buf ^ 1, k0 + 32);

    v16bf afr[2], bfr[4];
#pragma unroll
    for (int mi = 0; mi < 2; ++mi)
      afr[mi] = load_frag_a(&As[buf][0], waveM * 32 + mi * 16, 32, 0, lane);
#pragma unroll
    for (int ni = 0; ni < 4; ++ni)
      bfr[ni] = load_frag_b(&Bs[buf][0], waveN * 64 + ni * 16, 32, 0, lane);
#pragma unroll
    for (int mi = 0; mi < 2; ++mi)
#pragma unroll
      for (int ni = 0; ni < 4; ++ni)
        acc[mi][ni] = wmma_bf16(afr[mi], bfr[ni], acc[mi][ni]);
  }

  const float* gate = mod + 2 * D_MODEL;
  int halfoff = (lane >> 4) * 8;
#pragma unroll
  for (int mi = 0; mi < 2; ++mi)
#pragma unroll
    for (int ni = 0; ni < 4; ++ni) {
      int col = n0 + waveN * 64 + ni * 16 + (lane & 15);
      float bv = bias[col];
      float gv = gate[col];
#pragma unroll
      for (int e = 0; e < 8; ++e) {
        int row = m0 + waveM * 32 + mi * 16 + e + halfoff;
        size_t idx = (size_t)row * D_MODEL + col;
        out[idx] = x[idx] + (acc[mi][ni][e] + bv) * gv;
      }
    }
}

// ---------------- host launcher ----------------
extern "C" void kernel_launch(void* const* d_in, const int* in_sizes, int n_in,
                              void* d_out, int out_size, void* d_ws, size_t ws_size,
                              hipStream_t stream) {
  const float* x    = (const float*)d_in[0];
  const float* vec  = (const float*)d_in[1];
  const float* fcos = (const float*)d_in[2];
  const float* fsin = (const float*)d_in[3];
  const float* w1   = (const float*)d_in[4];
  const float* b1   = (const float*)d_in[5];
  const float* w2   = (const float*)d_in[6];
  const float* b2   = (const float*)d_in[7];
  const float* qw   = (const float*)d_in[8];
  const float* kw   = (const float*)d_in[9];
  const float* mw   = (const float*)d_in[10];
  const float* mb   = (const float*)d_in[11];

  char* ws = (char*)d_ws;
  size_t off = 0;
  float* mod   = (float*)(ws + off); off += (size_t)MOD_DIM * 4;
  off = (off + 255) & ~(size_t)255;
  bf16* xm     = (bf16*)(ws + off);  off += (size_t)S_LEN * D_MODEL * 2;        // 14 MB
  float* projq = (float*)(ws + off); off += (size_t)S_LEN * QKV_DIM * 4;        // 85 MB
  bf16* qh     = (bf16*)(ws + off);  off += (size_t)NHEAD * S_LEN * HDIM * 2;   // 14 MB
  bf16* kh     = (bf16*)(ws + off);  off += (size_t)NHEAD * S_LEN * HDIM * 2;   // 14 MB
  bf16* vt     = (bf16*)(ws + off);  off += (size_t)NHEAD * S_LEN * HDIM * 2;   // 14 MB
  bf16* a2     = (bf16*)(ws + off);  off += (size_t)S_LEN * K2_DIM * 2;         // 71 MB
  bf16* w1t    = (bf16*)(ws + off);  off += (size_t)PROJ_DIM * D_MODEL * 2;     // 132 MB
  bf16* w2t    = (bf16*)(ws + off);  off += (size_t)D_MODEL * K2_DIM * 2;       // 94 MB
  (void)ws_size; (void)in_sizes; (void)n_in; (void)out_size;

  // weight convert+transpose: w1[D][PROJ] -> w1t[PROJ][D]; w2[K2][D] -> w2t[D][K2]
  convt_kernel<<<dim3(PROJ_DIM / 32, D_MODEL / 32), 256, 0, stream>>>(
      w1, w1t, D_MODEL, PROJ_DIM);
  convt_kernel<<<dim3(D_MODEL / 32, K2_DIM / 32), 256, 0, stream>>>(
      w2, w2t, K2_DIM, D_MODEL);

  mod_gemv_kernel<<<MOD_DIM / 256, 256, 0, stream>>>(vec, mw, mb, mod);
  ln_mod_kernel<<<S_LEN, 256, 0, stream>>>(x, mod, xm);
  gemm1_kernel<<<dim3(PROJ_DIM / 128, S_LEN / 128), 256, 0, stream>>>(
      xm, w1t, b1, projq, a2);
  qkv_prep_kernel<<<dim3(S_LEN, NHEAD), 128, 0, stream>>>(
      projq, qw, kw, fcos, fsin, qh, kh, vt);
  attn_kernel<<<dim3(S_LEN / 128, NHEAD), 256, 0, stream>>>(qh, kh, vt, a2);
  gemm2_kernel<<<dim3(D_MODEL / 128, S_LEN / 128), 256, 0, stream>>>(
      a2, w2t, b2, mod, x, (float*)d_out);
}
